// EquivariantInterface_64338610094921
// MI455X (gfx1250) — matrix operand order repaired and checked by reading, not compiled, not run on hardware
//
#include <hip/hip_runtime.h>
#include <math.h>

typedef _Float16 h16;
typedef __attribute__((ext_vector_type(8)))  _Float16 v8h;
typedef __attribute__((ext_vector_type(16))) _Float16 v16h;
typedef __attribute__((ext_vector_type(8)))  float    v8f;

#define CH 96
#define K1PAD 640
#define D_IN 620
#define M_TOP 200
#define ROWS_PER_WG 128
#define THRESH 0.1f

// workspace layout (bytes)
#define OFF_FEAT 0u
#define SZ_FEAT  (32768u * 640u * 2u)            // 40 MiB f16 features
#define OFF_W1T  (OFF_FEAT + SZ_FEAT)
#define OFF_W2T  (OFF_W1T + 96u * 640u * 2u)
#define OFF_W3T  (OFF_W2T + 96u * 96u * 2u)
#define OFF_W4T  (OFF_W3T + 96u * 96u * 2u)

// ---------------------------------------------------------------------------
// Weight prep: fp32 [K][N] -> fp16 transposed [N][Kpad] (WMMA B layout wants
// contiguous K per output column).
// ---------------------------------------------------------------------------
__global__ __launch_bounds__(256) void prep_weights(
    const float* __restrict__ W1, const float* __restrict__ W2,
    const float* __restrict__ W3, const float* __restrict__ W4,
    h16* __restrict__ w1t, h16* __restrict__ w2t,
    h16* __restrict__ w3t, h16* __restrict__ w4t) {
  int tid = blockIdx.x * blockDim.x + threadIdx.x;
  if (tid < 96 * 640) {
    int n = tid / 640, k = tid % 640;
    w1t[n * 640 + k] = (k < D_IN) ? (h16)W1[k * 96 + n] : (h16)0.0f;
  } else if (tid < 96 * 640 + 96 * 96) {
    int t = tid - 96 * 640;
    int n = t / 96, k = t % 96;
    w2t[n * 96 + k] = (h16)W2[k * 96 + n];
  } else if (tid < 96 * 640 + 2 * 96 * 96) {
    int t = tid - 96 * 640 - 96 * 96;
    int n = t / 96, k = t % 96;
    w3t[n * 96 + k] = (h16)W3[k * 96 + n];
  } else if (tid < 96 * 640 + 2 * 96 * 96 + 16 * 96) {
    int t = tid - 96 * 640 - 2 * 96 * 96;
    int n = t / 96, k = t % 96;                  // N padded 4 -> 16
    w4t[n * 96 + k] = (n < 4) ? (h16)W4[k * 4 + n] : (h16)0.0f;
  }
}

// ---------------------------------------------------------------------------
// Per-image stable top-200 via in-LDS bitonic sort of 1024 uint64 keys.
// key = (float_bits(v) << 10) | (1023 - pixel_idx); v in [0,1) so the IEEE
// bit pattern is order-preserving; the tail makes the descending sort stable
// (smaller index first among equal values), matching argsort(-v) exactly.
// Pad slots get key 0 (any real key has tail >= 240 > 0 -> sorts last).
// feat layout: [0:200) intensity, [200:600) (cx,cy) pairs, [600:620) Z, pad 0.
// ---------------------------------------------------------------------------
__global__ __launch_bounds__(256) void build_feat(
    const float* __restrict__ images, const float* __restrict__ angles,
    h16* __restrict__ feat) {
  __shared__ unsigned long long keys[1024];
  const int b = blockIdx.x;
  const float* img = images + (size_t)b * 784;

  for (int i = threadIdx.x; i < 1024; i += 256) {
    unsigned long long k = 0ull;
    if (i < 784) {
      unsigned vb = __float_as_uint(img[i]);
      k = ((unsigned long long)vb << 10) | (unsigned long long)(1023 - i);
    }
    keys[i] = k;
  }
  __syncthreads();

  // bitonic sort, overall descending
  for (unsigned k = 2; k <= 1024; k <<= 1) {
    for (unsigned j = k >> 1; j > 0; j >>= 1) {
      for (unsigned i = threadIdx.x; i < 1024; i += 256) {
        unsigned ixj = i ^ j;
        if (ixj > i) {
          unsigned long long a = keys[i], c = keys[ixj];
          bool up = ((i & k) == 0);        // descending sub-run
          bool sw = up ? (a < c) : (a > c);
          if (sw) { keys[i] = c; keys[ixj] = a; }
        }
      }
      __syncthreads();
    }
  }

  h16* frow = feat + (size_t)b * K1PAD;
  if (threadIdx.x < M_TOP) {
    unsigned long long kk = keys[threadIdx.x];   // rank = threadIdx.x
    int p = 1023 - (int)(kk & 1023ull);
    float v = __uint_as_float((unsigned)(kk >> 10));
    int r = p / 28, c = p % 28;
    float cx = (c < 14) ? (float)(c - 14) : (float)(c - 13);
    float cy = (r < 14) ? (float)(14 - r) : (float)(13 - r);
    bool msk = v < THRESH;
    frow[threadIdx.x]                  = msk ? (h16)0.0f : (h16)v;
    frow[M_TOP + 2 * threadIdx.x]      = msk ? (h16)0.0f : (h16)cx;
    frow[M_TOP + 2 * threadIdx.x + 1]  = msk ? (h16)0.0f : (h16)cy;
  } else if (threadIdx.x < M_TOP + 10) {
    int d = threadIdx.x - M_TOP;
    float a = angles[(size_t)b * 10 + d];
    frow[600 + 2 * d]     = (h16)cosf(a);
    frow[600 + 2 * d + 1] = (h16)sinf(a);
  } else if (threadIdx.x < M_TOP + 10 + 20) {
    frow[620 + (threadIdx.x - (M_TOP + 10))] = (h16)0.0f;  // K pad 620..639
  }
}

// ---------------------------------------------------------------------------
// WMMA fragment loaders (16-bit A 16x32 / B 32x16 ISA layouts, wave32).
// A lane<16: K k0..k0+7 and k0+16..k0+23; lane>=16: +8 on both halves.
// B lane<16: K k0..k0+7 and k0+16..k0+23; lane>=16: +8 on both halves (column = lane&15).
// ---------------------------------------------------------------------------
__device__ __forceinline__ v16h load_a_frag(const h16* rowp, int k0, int halfsel) {
  v8h lo = *(const v8h*)(rowp + k0 + halfsel * 8);
  v8h hi = *(const v8h*)(rowp + k0 + 16 + halfsel * 8);
  v16h r;
#pragma unroll
  for (int i = 0; i < 8; i++) { r[i] = lo[i]; r[8 + i] = hi[i]; }
  return r;
}
__device__ __forceinline__ v16h load_b_frag(const h16* colp, int k0, int halfsel) {
  v8h lo = *(const v8h*)(colp + k0 + halfsel * 8);
  v8h hi = *(const v8h*)(colp + k0 + 16 + halfsel * 8);
  v16h r;
#pragma unroll
  for (int i = 0; i < 8; i++) { r[i] = lo[i]; r[8 + i] = hi[i]; }
  return r;
}
__device__ __forceinline__ v8f wmma_f16(v16h a, v16h b, v8f c) {
  return __builtin_amdgcn_wmma_f32_16x16x32_f16(false, a, false, b, (short)0, c,
                                                false, false);
}

// Software-pipelined K-loop: fragments for chunk k0+32 are fetched before the
// WMMAs consuming chunk k0 issue, so loadcnt waits overlap matrix math.
template <int KLEN, int NTILES, bool PREFETCH>
__device__ __forceinline__ void gemm_acc(const h16* __restrict__ arow,
                                         const h16* __restrict__ wt,
                                         int l15, int halfsel, v8f* acc) {
  v16h af = load_a_frag(arow, 0, halfsel);
  v16h bf[NTILES];
#pragma unroll
  for (int t = 0; t < NTILES; t++)
    bf[t] = load_b_frag(wt + (t * 16 + l15) * KLEN, 0, halfsel);
#pragma unroll
  for (int k0 = 0; k0 < KLEN; k0 += 32) {
    v16h afn = af;
    v16h bfn[NTILES];
#pragma unroll
    for (int t = 0; t < NTILES; t++) bfn[t] = bf[t];
    if (k0 + 32 < KLEN) {
      afn = load_a_frag(arow, k0 + 32, halfsel);
#pragma unroll
      for (int t = 0; t < NTILES; t++)
        bfn[t] = load_b_frag(wt + (t * 16 + l15) * KLEN, k0 + 32, halfsel);
    }
    if (PREFETCH) __builtin_prefetch(arow + k0 + 256, 0, 0);  // global_prefetch_b8
#pragma unroll
    for (int t = 0; t < NTILES; t++) acc[t] = wmma_f16(af, bf[t], acc[t]);
    af = afn;
#pragma unroll
    for (int t = 0; t < NTILES; t++) bf[t] = bfn[t];
  }
}

// ---------------------------------------------------------------------------
// Fused 4-layer MLP + Gram-Schmidt. 256 thr = 8 waves; wave w owns rows
// [16w,16w+16) of a 128-row block; 6 N-tiles cover CH=96.
// ---------------------------------------------------------------------------
__global__ __launch_bounds__(256) void mlp_kernel(
    const h16* __restrict__ feat, const h16* __restrict__ w1t,
    const h16* __restrict__ w2t, const h16* __restrict__ w3t,
    const h16* __restrict__ w4t, const float* __restrict__ b1,
    const float* __restrict__ b2, const float* __restrict__ b3,
    const float* __restrict__ b4, float* __restrict__ out) {
  __shared__ __align__(16) h16 hbuf[2][ROWS_PER_WG][CH];
  __shared__ float scr[8][16][4];

  const int lane = threadIdx.x & 31;
  const int wave = threadIdx.x >> 5;
  const int halfsel = lane >> 4;      // 0: lanes 0-15, 1: lanes 16-31
  const int l15 = lane & 15;
  const int mAdd = halfsel * 8;       // D rows 8..15 live in lanes 16-31
  const size_t rowGlobal = (size_t)blockIdx.x * ROWS_PER_WG + 16 * wave + l15;

  // ---- layer 1: feat(K=640) x W1t -> hbuf[0], bias+relu
  v8f acc[6];
#pragma unroll
  for (int t = 0; t < 6; t++) acc[t] = (v8f){0.f, 0.f, 0.f, 0.f, 0.f, 0.f, 0.f, 0.f};
  gemm_acc<K1PAD, 6, true>(feat + rowGlobal * K1PAD, w1t, l15, halfsel, acc);
#pragma unroll
  for (int t = 0; t < 6; t++) {
    float bb = b1[t * 16 + l15];
#pragma unroll
    for (int r = 0; r < 8; r++) {
      float v = acc[t][r] + bb;
      v = v > 0.f ? v : 0.f;
      hbuf[0][16 * wave + r + mAdd][t * 16 + l15] = (h16)v;
    }
  }
  __syncthreads();

  // ---- layers 2,3: hbuf[src](K=96) -> hbuf[dst], bias+relu
  const h16* wts[2] = {w2t, w3t};
  const float* bss[2] = {b2, b3};
#pragma unroll
  for (int li = 0; li < 2; li++) {
    const int src = li, dst = li ^ 1;
    v8f a2[6];
#pragma unroll
    for (int t = 0; t < 6; t++) a2[t] = (v8f){0.f, 0.f, 0.f, 0.f, 0.f, 0.f, 0.f, 0.f};
    gemm_acc<CH, 6, false>(&hbuf[src][16 * wave + l15][0], wts[li], l15, halfsel, a2);
#pragma unroll
    for (int t = 0; t < 6; t++) {
      float bb = bss[li][t * 16 + l15];
#pragma unroll
      for (int r = 0; r < 8; r++) {
        float v = a2[t][r] + bb;
        v = v > 0.f ? v : 0.f;
        hbuf[dst][16 * wave + r + mAdd][t * 16 + l15] = (h16)v;
      }
    }
    __syncthreads();
  }

  // ---- layer 4: hbuf[0](K=96) x W4t (N padded to 16; cols 0..3 valid)
  {
    v8f a4[1] = {(v8f){0.f, 0.f, 0.f, 0.f, 0.f, 0.f, 0.f, 0.f}};
    gemm_acc<CH, 1, false>(&hbuf[0][16 * wave + l15][0], w4t, l15, halfsel, a4);
    if (l15 < 4) {
      float bb = b4[l15];
#pragma unroll
      for (int r = 0; r < 8; r++) scr[wave][r + mAdd][l15] = a4[0][r] + bb;
    }
  }
  __syncthreads();

  // ---- per-row 2x2 Gram-Schmidt + det sign (lanes 0..15, one row each)
  if (lane < 16) {
    float o0 = scr[wave][lane][0], o1 = scr[wave][lane][1];
    float o2 = scr[wave][lane][2], o3 = scr[wave][lane][3];
    // gs = out.reshape(2,2).T ; c0 = (o0,o1), c1 = (o2,o3)
    float inv0 = rsqrtf(o0 * o0 + o1 * o1);
    float e0x = o0 * inv0, e0y = o1 * inv0;
    float d = e0x * o2 + e0y * o3;
    float u1x = o2 - d * e0x, u1y = o3 - d * e0y;
    float inv1 = rsqrtf(u1x * u1x + u1y * u1y);
    float e1x = u1x * inv1, e1y = u1y * inv1;
    float det = e0x * e1y - e1x * e0y;
    size_t row = (size_t)blockIdx.x * ROWS_PER_WG + 16 * wave + lane;
    out[row * 4 + 0] = e0x * det;   // q[0][0]*sign0
    out[row * 4 + 1] = e1x;         // q[0][1]
    out[row * 4 + 2] = e0y * det;   // q[1][0]*sign0
    out[row * 4 + 3] = e1y;         // q[1][1]
  }
}

extern "C" void kernel_launch(void* const* d_in, const int* in_sizes, int n_in,
                              void* d_out, int out_size, void* d_ws, size_t ws_size,
                              hipStream_t stream) {
  const float* images = (const float*)d_in[0];
  const float* angles = (const float*)d_in[1];
  const float* W1 = (const float*)d_in[2];
  const float* b1 = (const float*)d_in[3];
  const float* W2 = (const float*)d_in[4];
  const float* b2 = (const float*)d_in[5];
  const float* W3 = (const float*)d_in[6];
  const float* b3 = (const float*)d_in[7];
  const float* W4 = (const float*)d_in[8];
  const float* b4 = (const float*)d_in[9];

  char* ws = (char*)d_ws;
  h16* feat = (h16*)(ws + OFF_FEAT);
  h16* w1t = (h16*)(ws + OFF_W1T);
  h16* w2t = (h16*)(ws + OFF_W2T);
  h16* w3t = (h16*)(ws + OFF_W3T);
  h16* w4t = (h16*)(ws + OFF_W4T);

  int Bn = in_sizes[0] / 784;  // 32768

  int prep_elems = 96 * 640 + 2 * 96 * 96 + 16 * 96;
  prep_weights<<<(prep_elems + 255) / 256, 256, 0, stream>>>(W1, W2, W3, W4,
                                                             w1t, w2t, w3t, w4t);
  build_feat<<<Bn, 256, 0, stream>>>(images, angles, feat);
  mlp_kernel<<<Bn / ROWS_PER_WG, 256, 0, stream>>>(feat, w1t, w2t, w3t, w4t,
                                                   b1, b2, b3, b4, (float*)d_out);
}
